// GraphTransGeoPlusPlus_78546361909460
// MI455X (gfx1250) — matrix operand order repaired and checked by reading, not compiled, not run on hardware
//
#include <hip/hip_runtime.h>

typedef __attribute__((ext_vector_type(16))) _Float16 v16h;
typedef __attribute__((ext_vector_type(8)))  _Float16 v8h;
typedef __attribute__((ext_vector_type(8)))  float    v8f;

#define LN_EPS 1e-5f

static __device__ inline v16h cat8(v8h lo, v8h hi) {
    return __builtin_shufflevector(lo, hi, 0,1,2,3,4,5,6,7,8,9,10,11,12,13,14,15);
}

// ---------------------------------------------------------------------------
// Degree / dinv kernels
// ---------------------------------------------------------------------------
__global__ void deg_init_kernel(float* __restrict__ deg, int n) {
    int i = blockIdx.x * 256 + threadIdx.x;
    if (i < n) deg[i] = 1.0f;  // self-loop
}

__global__ void deg_edge_kernel(const int* __restrict__ dst, float* __restrict__ deg, int nE) {
    int e = blockIdx.x * 256 + threadIdx.x;
    if (e < nE) atomicAdd(&deg[dst[e]], 1.0f);
}

__global__ void deg_rsqrt_kernel(float* __restrict__ deg, int n) {
    int i = blockIdx.x * 256 + threadIdx.x;
    if (i < n) deg[i] = rsqrtf(deg[i]);  // deg -> dinv in place
}

// ---------------------------------------------------------------------------
// f32 -> f16 conversion (activations for WMMA A-matrix)
// ---------------------------------------------------------------------------
__global__ void f32_to_f16_kernel(const float* __restrict__ in, _Float16* __restrict__ out, int n) {
    int i = blockIdx.x * 256 + threadIdx.x;
    if (i < n) out[i] = (_Float16)in[i];
}

// ---------------------------------------------------------------------------
// WMMA GEMM: C[n_rows,128] = (relu?)(A16[n_rows,K] * Wf[K,128] + bias)
// 8 waves/block; wave w computes rows [blk*128 + w*16, +16) x all 128 cols.
// Weights are converted f32->f16 and pre-swizzled into LDS in exact
// B-fragment order so each fragment is 2 contiguous ds_load_b128 per lane.
// ---------------------------------------------------------------------------
template <int K>
__global__ __launch_bounds__(256) void gemm_wmma_kernel(
    const _Float16* __restrict__ A16, const float* __restrict__ Wf,
    const float* __restrict__ bias, float* __restrict__ C,
    int n_rows, int do_relu)
{
    constexpr int KT = K / 32;                 // k-tiles of 32
    __shared__ _Float16 wlds[K * 128];         // swizzled B fragments

    const int tid = threadIdx.x;
    // Fill LDS: source-coalesced read of Wf (row-major [K][128]),
    // scatter into fragment layout: frag[kt][ct][lane][i], where within a
    // 32x16 tile: Kloc = (i/8)*16 + (lane/16)*8 + i%8, Nloc = lane%16.
    for (int idx = tid; idx < K * 128; idx += 256) {
        int K_idx = idx >> 7, N_idx = idx & 127;
        int kt = K_idx >> 5, klo = K_idx & 31;
        int ihi = (klo >> 3) & 1, i = (klo & 7) + ((klo >> 4) << 3);
        int ct = N_idx >> 4, lanelo = N_idx & 15;
        int lane = ihi * 16 + lanelo;
        wlds[(((kt * 8 + ct) * 32) + lane) * 16 + i] = (_Float16)Wf[idx];
    }
    __syncthreads();

    const int wave = tid >> 5;
    const int lane = tid & 31;
    const int row_base = (blockIdx.x * 8 + wave) * 16;
    if (row_base >= n_rows) return;            // n_rows % 16 == 0: tiles exact

    const int m    = row_base + (lane & 15);
    const int koff = (lane >> 4) * 8;          // A layout: hi half-wave holds K+8

    v8f acc[8];
    #pragma unroll
    for (int ct = 0; ct < 8; ++ct) {
        float bv = bias ? bias[ct * 16 + (lane & 15)] : 0.0f;
        #pragma unroll
        for (int r = 0; r < 8; ++r) acc[ct][r] = bv;
    }

    const _Float16* arow = A16 + (size_t)m * K;
    #pragma unroll
    for (int kt = 0; kt < KT; ++kt) {
        // A fragment (16x32 f16): lanes<16 hold K = {0..7,16..23}+kt*32,
        // lanes>=16 hold K = {8..15,24..31}+kt*32, row m fixed per lane.
        v8h alo = *(const v8h*)(arow + kt * 32 + koff);
        v8h ahi = *(const v8h*)(arow + kt * 32 + 16 + koff);
        v16h a = cat8(alo, ahi);
        #pragma unroll
        for (int ct = 0; ct < 8; ++ct) {
            const _Float16* bp = &wlds[(((kt * 8 + ct) * 32) + lane) * 16];
            v16h b = cat8(*(const v8h*)bp, *(const v8h*)(bp + 8));
            acc[ct] = __builtin_amdgcn_wmma_f32_16x16x32_f16(
                false, a, false, b, (short)0, acc[ct], false, false);
        }
    }

    // C/D layout: VGPR r, lane l -> M = row_base + (l/16)*8 + r, N = ct*16 + l%16
    const int mm0 = row_base + (lane >> 4) * 8;
    const int nn  = lane & 15;
    #pragma unroll
    for (int ct = 0; ct < 8; ++ct) {
        #pragma unroll
        for (int r = 0; r < 8; ++r) {
            float v = acc[ct][r];
            if (do_relu) v = fmaxf(v, 0.0f);
            C[(size_t)(mm0 + r) * 128 + ct * 16 + nn] = v;
        }
    }
}

// ---------------------------------------------------------------------------
// agg[v] = hw[v] * dinv[v]^2 + bias   (self-loop contribution + conv bias)
// wave per node, float4 per lane (128 cols / 32 lanes)
// ---------------------------------------------------------------------------
__global__ __launch_bounds__(256) void agg_init_kernel(
    const float* __restrict__ hw, const float* __restrict__ dinv,
    const float* __restrict__ bias, float* __restrict__ agg, int n)
{
    int gid = blockIdx.x * 256 + threadIdx.x;
    int node = gid >> 5, lane = gid & 31;
    if (node >= n) return;
    float w = dinv[node]; w = w * w;
    float4 v = *(const float4*)(hw + (size_t)node * 128 + lane * 4);
    float4 b = *(const float4*)(bias + lane * 4);
    float4 o = { v.x * w + b.x, v.y * w + b.y, v.z * w + b.z, v.w * w + b.w };
    *(float4*)(agg + (size_t)node * 128 + lane * 4) = o;
}

// ---------------------------------------------------------------------------
// Edge scatter: agg[dst] += hw[src] * dinv[src]*dinv[dst]. Wave per edge.
// ---------------------------------------------------------------------------
__global__ __launch_bounds__(256) void edge_scatter_kernel(
    const int* __restrict__ src, const int* __restrict__ dst,
    const float* __restrict__ dinv, const float* __restrict__ hw,
    float* __restrict__ agg, int nE)
{
    long long gid = (long long)blockIdx.x * 256 + threadIdx.x;
    int e = (int)(gid >> 5);
    int lane = (int)(gid & 31);
    if (e >= nE) return;
    int s = src[e], d = dst[e];
    float norm = dinv[s] * dinv[d];
    float4 v = *(const float4*)(hw + (size_t)s * 128 + lane * 4);
    float* out = agg + (size_t)d * 128 + lane * 4;
    atomicAdd(out + 0, v.x * norm);
    atomicAdd(out + 1, v.y * norm);
    atomicAdd(out + 2, v.z * norm);
    atomicAdd(out + 3, v.w * norm);
}

// ---------------------------------------------------------------------------
// h = relu(LayerNorm(agg + h) * g + be)   (in place on h). Wave per node.
// ---------------------------------------------------------------------------
__global__ __launch_bounds__(256) void ln_relu_kernel(
    const float* __restrict__ agg, float* __restrict__ h,
    const float* __restrict__ g, const float* __restrict__ be, int n)
{
    int gid = blockIdx.x * 256 + threadIdx.x;
    int node = gid >> 5, lane = gid & 31;
    if (node >= n) return;
    float4 a = *(const float4*)(agg + (size_t)node * 128 + lane * 4);
    float4 r = *(const float4*)(h + (size_t)node * 128 + lane * 4);
    float4 x = { a.x + r.x, a.y + r.y, a.z + r.z, a.w + r.w };

    float s = x.x + x.y + x.z + x.w;
    #pragma unroll
    for (int off = 16; off > 0; off >>= 1) s += __shfl_xor(s, off, 32);
    float mu = s * (1.0f / 128.0f);

    float dx0 = x.x - mu, dx1 = x.y - mu, dx2 = x.z - mu, dx3 = x.w - mu;
    float q = dx0 * dx0 + dx1 * dx1 + dx2 * dx2 + dx3 * dx3;
    #pragma unroll
    for (int off = 16; off > 0; off >>= 1) q += __shfl_xor(q, off, 32);
    float inv = rsqrtf(q * (1.0f / 128.0f) + LN_EPS);

    float4 gg = *(const float4*)(g + lane * 4);
    float4 bb = *(const float4*)(be + lane * 4);
    float4 o;
    o.x = fmaxf(dx0 * inv * gg.x + bb.x, 0.0f);
    o.y = fmaxf(dx1 * inv * gg.y + bb.y, 0.0f);
    o.z = fmaxf(dx2 * inv * gg.z + bb.z, 0.0f);
    o.w = fmaxf(dx3 * inv * gg.w + bb.w, 0.0f);
    *(float4*)(h + (size_t)node * 128 + lane * 4) = o;
}

// ---------------------------------------------------------------------------
// out[n,2] = h[n,:] @ W_out[128,2] + b_out. Wave per node.
// ---------------------------------------------------------------------------
__global__ __launch_bounds__(256) void out_proj_kernel(
    const float* __restrict__ h, const float* __restrict__ Wout,
    const float* __restrict__ bout, float* __restrict__ out, int n)
{
    int gid = blockIdx.x * 256 + threadIdx.x;
    int node = gid >> 5, lane = gid & 31;
    if (node >= n) return;
    float4 v = *(const float4*)(h + (size_t)node * 128 + lane * 4);
    int k = lane * 4;                       // rows k..k+3 of W_out [128][2]
    float4 w01 = *(const float4*)(Wout + k * 2);
    float4 w23 = *(const float4*)(Wout + k * 2 + 4);
    float a0 = v.x * w01.x + v.y * w01.z + v.z * w23.x + v.w * w23.z;
    float a1 = v.x * w01.y + v.y * w01.w + v.z * w23.y + v.w * w23.w;
    #pragma unroll
    for (int off = 16; off > 0; off >>= 1) {
        a0 += __shfl_xor(a0, off, 32);
        a1 += __shfl_xor(a1, off, 32);
    }
    if (lane == 0) {
        out[(size_t)node * 2 + 0] = a0 + bout[0];
        out[(size_t)node * 2 + 1] = a1 + bout[1];
    }
}

// ---------------------------------------------------------------------------
extern "C" void kernel_launch(void* const* d_in, const int* in_sizes, int n_in,
                              void* d_out, int out_size, void* d_ws, size_t ws_size,
                              hipStream_t stream) {
    const float* x     = (const float*)d_in[0];
    const int*   ei    = (const int*)d_in[1];     // [2,E] (src row, dst row)
    const float* W_in  = (const float*)d_in[2];
    const float* b_in  = (const float*)d_in[3];
    const float* W_out = (const float*)d_in[4];
    const float* b_out = (const float*)d_in[5];
    const float* Wc[3] = { (const float*)d_in[6],  (const float*)d_in[10], (const float*)d_in[14] };
    const float* bc[3] = { (const float*)d_in[7],  (const float*)d_in[11], (const float*)d_in[15] };
    const float* g[3]  = { (const float*)d_in[8],  (const float*)d_in[12], (const float*)d_in[16] };
    const float* be[3] = { (const float*)d_in[9],  (const float*)d_in[13], (const float*)d_in[17] };

    const int N = in_sizes[0] / 64;               // 100000
    const int E = in_sizes[1] / 2;                // 1600000
    float* out = (float*)d_out;

    // Workspace carve-up (256B-aligned)
    char* ws = (char*)d_ws;
    size_t off = 0;
    auto carve = [&](size_t bytes) -> void* {
        void* p = ws + off;
        off += (bytes + 255) & ~(size_t)255;
        return p;
    };
    float*    dinv = (float*)   carve((size_t)N * 4);
    float*    h    = (float*)   carve((size_t)N * 128 * 4);
    float*    hw   = (float*)   carve((size_t)N * 128 * 4);
    float*    agg  = (float*)   carve((size_t)N * 128 * 4);
    _Float16* h16  = (_Float16*)carve((size_t)N * 128 * 2);

    const int*  src = ei;
    const int*  dst = ei + E;

    const int nvB  = (N + 255) / 256;                     // node-elementwise blocks
    const int neB  = (E + 255) / 256;                     // edge blocks
    const int rowB = ((N + 15) / 16 + 7) / 8;             // GEMM blocks: 8 wave-tiles each
    const int nwB  = ((N * 32) + 255) / 256;              // wave-per-node blocks
    const int ewB  = (int)(((long long)E * 32 + 255) / 256); // wave-per-edge blocks

    // --- degrees -> dinv ---
    deg_init_kernel<<<nvB, 256, 0, stream>>>(dinv, N);
    deg_edge_kernel<<<neB, 256, 0, stream>>>(dst, dinv, E);
    deg_rsqrt_kernel<<<nvB, 256, 0, stream>>>(dinv, N);

    // --- input layer: h = relu(x @ W_in + b_in) ---
    f32_to_f16_kernel<<<(N * 64 + 255) / 256, 256, 0, stream>>>(x, h16, N * 64);
    gemm_wmma_kernel<64><<<rowB, 256, 0, stream>>>(h16, W_in, b_in, h, N, 1);

    // --- 3 GCN conv layers ---
    for (int l = 0; l < 3; ++l) {
        f32_to_f16_kernel<<<(N * 128 + 255) / 256, 256, 0, stream>>>(h, h16, N * 128);
        gemm_wmma_kernel<128><<<rowB, 256, 0, stream>>>(h16, Wc[l], nullptr, hw, N, 0);
        agg_init_kernel<<<nwB, 256, 0, stream>>>(hw, dinv, bc[l], agg, N);
        edge_scatter_kernel<<<ewB, 256, 0, stream>>>(src, dst, dinv, hw, agg, E);
        ln_relu_kernel<<<nwB, 256, 0, stream>>>(agg, h, g[l], be[l], N);
    }

    // --- output projection ---
    out_proj_kernel<<<nwB, 256, 0, stream>>>(h, W_out, b_out, out, N);
}
